// SamePadConvAdvanced_34368328302984
// MI455X (gfx1250) — matrix operand order repaired and checked by reading, not compiled, not run on hardware
//
#include <hip/hip_runtime.h>
#include <hip/hip_bf16.h>
#include <math.h>

#define O_  64
#define I_  64
#define KT  3
#define NH_ 64
#define M_  32
#define T_  8192
#define B_  64
#define QD  320
#define KK  192   // I_*KT contraction length
#define TT  128   // time tile
#define AST 200   // As LDS row stride (bf16)
#define BST 72    // Bs LDS row stride (bf16)

typedef __attribute__((ext_vector_type(16))) __bf16 bf16x16;
typedef __attribute__((ext_vector_type(8)))  __bf16 bf16x8;
typedef __attribute__((ext_vector_type(8)))  float  f32x8;

__device__ __forceinline__ float sigmoidf_(float x){ return 1.0f/(1.0f+__expf(-x)); }
__device__ __forceinline__ float siluf_(float x){ return x*sigmoidf_(x); }

// ---------------------------------------------------------------------------
// Stage 1: hypernetwork -> modulated bf16 weight matrix A[o][ktap*64+i] and
// per-channel bias2[o] (fq folded into both).
// ---------------------------------------------------------------------------
__global__ __launch_bounds__(256)
void hyper_prep_kernel(const float* __restrict__ grads, const float* __restrict__ q_ema,
                       const float* __restrict__ W,     const float* __restrict__ conv_w,
                       const float* __restrict__ conv_b,
                       const float* __restrict__ ctrl_w, const float* __restrict__ ctrl_b,
                       const float* __restrict__ cw_w,   const float* __restrict__ cw_b,
                       const float* __restrict__ cb_w,   const float* __restrict__ cb_b,
                       const float* __restrict__ cf_w,   const float* __restrict__ cf_b,
                       const float* __restrict__ tau_w1, const float* __restrict__ tau_b1,
                       const float* __restrict__ tau_w2, const float* __restrict__ tau_b2,
                       const int* __restrict__ trigger,
                       __bf16* __restrict__ Aw, float* __restrict__ bias2)
{
  __shared__ float rep[O_][NH_];
  __shared__ float q[QD];
  __shared__ float tauv[O_];
  __shared__ float att[M_];
  __shared__ float tau_s;
  __shared__ float topv[3];
  __shared__ int   topi[3];
  const int tid = threadIdx.x;

  // rep = silu(g @ ctrl_w^T + ctrl_b), g = grads.reshape(64,192)
  for (int idx = tid; idx < O_*NH_; idx += 256) {
    int o = idx >> 6, h = idx & 63;
    float s = ctrl_b[h];
    const float* gp = grads  + o*KK;
    const float* cp = ctrl_w + h*KK;
    for (int j = 0; j < KK; ++j) s += gp[j]*cp[j];
    rep[o][h] = siluf_(s);
  }
  __syncthreads();

  if (tid < O_) {
    const int o = tid;
    float w0=cw_b[0], w1=cw_b[1], w2=cw_b[2], bb=cb_b[0], ff=cf_b[0];
    for (int h = 0; h < NH_; ++h) {
      float r = rep[o][h];
      w0 += r*cw_w[0*NH_+h]; w1 += r*cw_w[1*NH_+h]; w2 += r*cw_w[2*NH_+h];
      bb += r*cb_w[h];       ff += r*cf_w[h];
    }
    q[o*3+0]=w0; q[o*3+1]=w1; q[o*3+2]=w2;
    q[192+o]=bb; q[256+o]=ff;
    float tacc = tau_b2[0];
    for (int j=0;j<16;++j){
      float hs = tau_b1[j];
      for (int h=0;h<NH_;++h) hs += rep[o][h]*tau_w1[j*NH_+h];
      tacc += siluf_(hs)*tau_w2[j];
    }
    tauv[o] = sigmoidf_(tacc);
  }
  __syncthreads();
  if (tid == 0) {
    float s=0.f; for (int o=0;o<O_;++o) s += tauv[o];
    tau_s = (s/(float)O_)*0.5f + 0.5f;
  }
  __syncthreads();

  // EMA blend (QD=320 > blockDim, so strided loop)
  for (int idx = tid; idx < QD; idx += 256) {
    float qv = q[idx];
    float qe = 0.3f*q_ema[idx] + 0.7f*qv;
    q[idx] = 0.5f*qv + 0.5f*qe;
  }
  __syncthreads();

  if (*trigger == 1) {                  // uniform branch
    if (tid < M_) {
      float s = 0.f;
      for (int i=0;i<QD;++i) s += q[i]*W[i*M_+tid];
      att[tid] = s*2.0f;                // /TEMP, TEMP=0.5
    }
    __syncthreads();
    if (tid == 0) {
      float mx = -1e30f;
      for (int m=0;m<M_;++m) mx = fmaxf(mx, att[m]);
      float den = 0.f;
      for (int m=0;m<M_;++m){ att[m] = __expf(att[m]-mx); den += att[m]; }
      float inv = 1.0f/den;
      for (int m=0;m<M_;++m) att[m]*=inv;
      for (int j=0;j<3;++j){
        float bv=-1e30f; int bi=0;
        for (int m=0;m<M_;++m) if (att[m] > bv){bv=att[m];bi=m;}
        topv[j]=bv; topi[j]=bi; att[bi]=-1e30f;
      }
    }
    __syncthreads();
    for (int idx = tid; idx < QD; idx += 256) {
      float oq = 0.f;
      for (int j=0;j<3;++j) oq += W[idx*M_+topi[j]]*topv[j];
      q[idx] = tau_s*q[idx] + (1.0f-tau_s)*oq;
    }
  }
  __syncthreads();

  // A[o][ktap*64+i] = conv_w[o,i,ktap] * wq[o,ktap] * fq[o]  (bf16)
  for (int idx = tid; idx < O_*KK; idx += 256) {
    int o = idx / KK;
    int r = idx - o*KK;
    int ktap = r >> 6;
    int i    = r & 63;
    float v = conv_w[(o*I_+i)*KT + ktap] * q[o*3+ktap] * q[256+o];
    Aw[idx] = (__bf16)v;
  }
  if (tid < O_) bias2[tid] = q[256+tid]*conv_b[tid]*q[192+tid];
}

// ---------------------------------------------------------------------------
// Stage 2: HBM-bound conv-as-GEMM. Per block: one batch b, 128 t positions,
// all 64 output channels. D(64x128) = A(64x192) x B(192x128), bf16 WMMA.
// x tile staged fp32 via global_load_async_to_lds_b128, converted to a
// transposed bf16 [t][i] tile in LDS.
// ---------------------------------------------------------------------------
__global__ __launch_bounds__(256)
void conv_wmma_kernel(const float* __restrict__ x, const __bf16* __restrict__ Aw,
                      const float* __restrict__ bias2, float* __restrict__ out)
{
  __shared__ float  xs[I_][132];       // fp32 bulk tile, cols 0..127 used
  __shared__ float  xlo[I_], xhi[I_];  // halo columns (t0-1, t0+128)
  __shared__ __bf16 Bs[130*BST];       // [tt][i], tt -> t_global = t0-1+tt
  __shared__ __bf16 As[O_*AST];        // [o][kk], padded rows
  __shared__ float  bias_s[O_];

  const int tid = threadIdx.x;
  const int t0  = blockIdx.x * TT;
  const int b   = blockIdx.y;
  const float* xb = x + ((size_t)b * I_) * T_ + t0;

  // ---- async bulk x tile: 64 rows x 512B, 16B per lane-op, 8 iterations ----
  {
    unsigned lds_base = (unsigned)(unsigned long long)(void*)&xs[0][0];
    unsigned long long gbase = (unsigned long long)(const void*)xb;
    #pragma unroll
    for (int it = 0; it < 8; ++it) {
      int chunk = it*256 + tid;                 // 0..2047
      int i = chunk >> 5;
      int c = chunk & 31;
      unsigned lds_off = lds_base + (unsigned)(i*132*4 + c*16);
      unsigned voff    = (unsigned)(i*T_*4 + c*16);
      asm volatile("global_load_async_to_lds_b128 %0, %1, %2"
                   :: "v"(lds_off), "v"(voff), "s"(gbase) : "memory");
    }
  }

  // ---- overlap: halos, weights, bias into LDS ----
  if (tid < I_) {
    xlo[tid] = (t0 > 0) ? xb[(size_t)tid*T_ - 1] : 0.0f;
  } else if (tid < 2*I_) {
    int i = tid - I_;
    xhi[i] = (t0 + TT < T_) ? xb[(size_t)i*T_ + TT] : 0.0f;
  }
  for (int idx = tid; idx < O_*KK; idx += 256) {
    int o = idx / KK;
    As[o*AST + (idx - o*KK)] = Aw[idx];
  }
  if (tid < O_) bias_s[tid] = bias2[tid];

  asm volatile("s_wait_asynccnt 0x0" ::: "memory");
  __syncthreads();

  // ---- fp32 -> bf16, transpose to [tt][i] with padded stride ----
  for (int idx = tid; idx < 130*I_; idx += 256) {
    int tt = idx >> 6;
    int i  = idx & 63;
    float v = (tt == 0) ? xlo[i] : ((tt == 129) ? xhi[i] : xs[i][tt-1]);
    Bs[tt*BST + i] = (__bf16)v;
  }
  __syncthreads();

  // ---- WMMA GEMM: 8 waves; wave -> (m0 = (w&3)*16, n0 = (w>>2)*64) ----
  const int w    = tid >> 5;
  const int lane = tid & 31;
  const int m0   = (w & 3) * 16;
  const int n0   = (w >> 2) * 64;
  const int lh   = lane >> 4;      // lane half
  const int ll   = lane & 15;

  f32x8 acc0 = {}, acc1 = {}, acc2 = {}, acc3 = {};

  const __bf16* aptr = &As[(m0 + ll)*AST];
  #pragma unroll
  for (int s = 0; s < 6; ++s) {
    // A fragment: lane holds M=m0+ll; elems 0..15 -> K=kb..kb+15
    int kbase = s*32 + lh*16;
    bf16x8 alo = *(const bf16x8*)(aptr + kbase);
    bf16x8 ahi = *(const bf16x8*)(aptr + kbase + 8);
    bf16x16 a;
    #pragma unroll
    for (int j = 0; j < 8; ++j) { a[j] = alo[j]; a[j+8] = ahi[j]; }

    // B fragments: lane holds N=n0+nt*16+ll; K chunk = s*32 + lh*16 (+j)
    int kb   = s*32 + lh*16;
    int ktap = kb >> 6;
    int i0   = kb & 63;
    #pragma unroll
    for (int nt = 0; nt < 4; ++nt) {
      int ttrow = n0 + nt*16 + ll + ktap;       // = n + ktap, t = t0-1+ttrow
      const __bf16* bp = &Bs[ttrow*BST + i0];
      bf16x8 blo = *(const bf16x8*)(bp);
      bf16x8 bhi = *(const bf16x8*)(bp + 8);
      bf16x16 bm;
      #pragma unroll
      for (int j = 0; j < 8; ++j) { bm[j] = blo[j]; bm[j+8] = bhi[j]; }
      f32x8& acc = (nt==0)?acc0:((nt==1)?acc1:((nt==2)?acc2:acc3));
      acc = __builtin_amdgcn_wmma_f32_16x16x32_bf16(false, a, false, bm,
                                                    (short)0, acc, false, false);
    }
  }

  // ---- store: D VGPR r -> M = m0 + r + 8*lh, N = ll ----
  float* ob = out + ((size_t)b * O_) * T_ + t0;
  #pragma unroll
  for (int nt = 0; nt < 4; ++nt) {
    const f32x8& acc = (nt==0)?acc0:((nt==1)?acc1:((nt==2)?acc2:acc3));
    int col = n0 + nt*16 + ll;
    #pragma unroll
    for (int r = 0; r < 8; ++r) {
      int m = m0 + r + lh*8;
      ob[(size_t)m * T_ + col] = acc[r] + bias_s[m];
    }
  }
}

// ---------------------------------------------------------------------------
extern "C" void kernel_launch(void* const* d_in, const int* in_sizes, int n_in,
                              void* d_out, int out_size, void* d_ws, size_t ws_size,
                              hipStream_t stream)
{
  const float* x      = (const float*)d_in[0];
  const float* grads  = (const float*)d_in[1];
  const float* q_ema  = (const float*)d_in[2];
  const float* W      = (const float*)d_in[3];
  const float* conv_w = (const float*)d_in[4];
  const float* conv_b = (const float*)d_in[5];
  const float* ctrl_w = (const float*)d_in[6];
  const float* ctrl_b = (const float*)d_in[7];
  const float* cw_w   = (const float*)d_in[8];
  const float* cw_b   = (const float*)d_in[9];
  const float* cb_w   = (const float*)d_in[10];
  const float* cb_b   = (const float*)d_in[11];
  const float* cf_w   = (const float*)d_in[12];
  const float* cf_b   = (const float*)d_in[13];
  const float* tau_w1 = (const float*)d_in[14];
  const float* tau_b1 = (const float*)d_in[15];
  const float* tau_w2 = (const float*)d_in[16];
  const float* tau_b2 = (const float*)d_in[17];
  const int*   trig   = (const int*)d_in[18];

  __bf16* Aw    = (__bf16*)d_ws;
  float*  bias2 = (float*)((char*)d_ws + (size_t)O_*KK*sizeof(__bf16));
  float*  out   = (float*)d_out;

  hyper_prep_kernel<<<dim3(1), dim3(256), 0, stream>>>(
      grads, q_ema, W, conv_w, conv_b, ctrl_w, ctrl_b, cw_w, cw_b,
      cb_w, cb_b, cf_w, cf_b, tau_w1, tau_b1, tau_w2, tau_b2, trig, Aw, bias2);

  conv_wmma_kernel<<<dim3(T_/TT, B_), dim3(256), 0, stream>>>(x, Aw, bias2, out);
}
